// RelationMultiHeadAttention_83777632076542
// MI455X (gfx1250) — matrix operand order repaired and checked by reading, not compiled, not run on hardware
//
#include <hip/hip_runtime.h>
#include <hip/hip_bf16.h>

#define B_ 2
#define L_ 256
#define E_ 512
#define H_ 8
#define D_ 64
#define TWOE_ 1024
#define SCALE_ 0.125f

typedef __attribute__((ext_vector_type(16))) __bf16 v16bf;
typedef __attribute__((ext_vector_type(8)))  __bf16 v8bf;
typedef __attribute__((ext_vector_type(8)))  float  v8f;
typedef __attribute__((ext_vector_type(4)))  float  v4f;

typedef unsigned int u32x4 __attribute__((ext_vector_type(4)));
typedef int          i32x4 __attribute__((ext_vector_type(4)));
typedef int          i32x8 __attribute__((ext_vector_type(8)));

#if __has_builtin(__builtin_amdgcn_tensor_load_to_lds)
#define USE_TDM 1
#else
#define USE_TDM 0
#endif

// ---------------------------------------------------------------- fp32->bf16
__global__ __launch_bounds__(256) void cvt_f32_bf16(const float* __restrict__ src,
                                                    __bf16* __restrict__ dst, int n) {
  int i = blockIdx.x * 256 + threadIdx.x;
  if (i < n) dst[i] = (__bf16)src[i];
}

// ------------------------------------------------- generic WMMA GEMM: Y = (A @ W^T + bias) * scale
template <bool A_F32>
__global__ __launch_bounds__(256) void gemm_bias(const void* __restrict__ Av,
                                                 const __bf16* __restrict__ W,
                                                 const float* __restrict__ bias,
                                                 float* __restrict__ Y,
                                                 int M, int N, int K, float scale) {
  const int tid  = threadIdx.x;
  const int wave = tid >> 5;
  const int lane = tid & 31;
  const int ln   = lane & 15;
  const int hl   = lane >> 4;
  const int mBase = blockIdx.x * 16;
  const int nWave = wave * 64;

  v8f acc[4] = {};
  for (int kc = 0; kc < K; kc += 32) {
    v16bf a;
    const int row = mBase + ln;
    if (A_F32) {
      const float* ap = (const float*)Av + (size_t)row * K + kc + hl * 16;
      v4f x0 = *(const v4f*)(ap);
      v4f x1 = *(const v4f*)(ap + 4);
      v4f x2 = *(const v4f*)(ap + 8);
      v4f x3 = *(const v4f*)(ap + 12);
#pragma unroll
      for (int e = 0; e < 4; ++e) {
        a[e] = (__bf16)x0[e]; a[4 + e] = (__bf16)x1[e];
        a[8 + e] = (__bf16)x2[e]; a[12 + e] = (__bf16)x3[e];
      }
    } else {
      const __bf16* ap = (const __bf16*)Av + (size_t)row * K + kc + hl * 16;
      v8bf lo = *(const v8bf*)(ap);
      v8bf hi = *(const v8bf*)(ap + 8);
#pragma unroll
      for (int e = 0; e < 8; ++e) { a[e] = lo[e]; a[8 + e] = hi[e]; }
    }
#pragma unroll
    for (int nb = 0; nb < 4; ++nb) {
      const int n = nWave + nb * 16 + ln;
      v16bf bf = *(const v16bf*)(W + (size_t)n * K + kc + hl * 16);
      acc[nb] = __builtin_amdgcn_wmma_f32_16x16x32_bf16(false, a, false, bf,
                                                        (short)0, acc[nb], false, false);
    }
  }
#pragma unroll
  for (int nb = 0; nb < 4; ++nb) {
    const int n = nWave + nb * 16 + ln;
    const float bv = bias ? bias[n] : 0.0f;
#pragma unroll
    for (int r = 0; r < 8; ++r) {
      const int m = mBase + r + hl * 8;
      Y[(size_t)m * N + n] = (acc[nb][r] + bv) * scale;
    }
  }
}

// ---------------------------------------------------------------- TDM helpers
__device__ __forceinline__ unsigned lds_off(const void* p) {
  return (unsigned)(unsigned long long)(uintptr_t)p;
}

#if USE_TDM
// DMA a [1024 rows x 32 cols] bf16 panel of Wr (row stride 512 elems) into LDS,
// padded to an 80-byte LDS row pitch (16 dwords data + 4 dwords pad).
__device__ __forceinline__ void tdm_load_wr_panel(const __bf16* gsrc, unsigned ldsByteOff) {
  u32x4 g0;
  g0[0] = 1u;                                    // count=1 (valid user descriptor)
  g0[1] = ldsByteOff;                            // lds_addr
  unsigned long long ga = (unsigned long long)(uintptr_t)gsrc;
  g0[2] = (unsigned)ga;                          // global_addr[31:0]
  g0[3] = (unsigned)(ga >> 32) | 0x80000000u;    // global_addr[56:32] | type=2
  i32x8 g1;
  g1[0] = (int)((1u << 16)      // data_size = 2 bytes
              | (1u << 20)      // pad_enable
              | (3u << 22)      // pad_interval: every 16 dwords
              | (3u << 25));    // pad_amount: 4 dwords
  g1[1] = (int)(512u << 16);    // tensor_dim0 = 512 (low 16 -> bits 63:48)
  g1[2] = (int)(1024u << 16);   // tensor_dim1 = 1024 (low 16 -> bits 95:80)
  g1[3] = (int)(32u << 16);     // tile_dim0 = 32 (bits 127:112)
  g1[4] = 1024;                 // tile_dim1 = 1024 (bits 143:128)
  g1[5] = 512;                  // tensor_dim0_stride = 512 (bits 207:160 low)
  g1[6] = 0;
  g1[7] = 0;
  i32x4 z4 = {0, 0, 0, 0};
  i32x8 z8 = {0, 0, 0, 0, 0, 0, 0, 0};
  __builtin_amdgcn_tensor_load_to_lds(g0, g1, z4, z4, z8, 0);
}
#endif

// ---------------------------------------------------------------- fused relation GEMM + logits
// Per block: (b, t, 32-row s-tile). rel[32x1024] = relation[b,t,sTile,:] @ Wr^T via bf16 WMMA.
// Wr panels streamed into double-buffered LDS by the Tensor Data Mover.
#define RELPITCH 1040                 // bf16 elems per staged rel row
#define BT_PITCH 80                   // bytes per 32-elem k-row in LDS B panel
#define BT_BYTES (1024 * BT_PITCH)    // 81920 bytes per panel
__global__ __launch_bounds__(256) void rel_attn_logits(const float* __restrict__ relation,
                                                       const __bf16* __restrict__ WrBf,
                                                       const float* __restrict__ qf,
                                                       const float* __restrict__ kf,
                                                       float* __restrict__ attn) {
  __shared__ __attribute__((aligned(128))) char smem[2 * BT_BYTES];  // 160 KB (CDNA5: 320 KB/WGP)

  const int tid  = threadIdx.x;
  const int wave = tid >> 5;
  const int lane = tid & 31;
  const int ln   = lane & 15;
  const int hl   = lane >> 4;
  const int wy   = wave >> 2;  // 0..1 : 16-row half of the 32-row tile
  const int wx   = wave & 3;   // 0..3 : 256-col quarter of the 1024 cols

  int bid = blockIdx.x;
  const int st = bid & 7;   bid >>= 3;
  const int t  = bid & 255; bid >>= 8;
  const int b  = bid;
  const int sBase = st * 32;

  const float* Abase = relation + (((size_t)b * L_ + t) * L_ + sBase) * E_;
  const int mRow = wy * 16 + ln;

  char* buf0 = smem;
  char* buf1 = smem + BT_BYTES;

#if USE_TDM
  if (wave == 0) tdm_load_wr_panel(WrBf, lds_off(buf0));
#endif

  v8f acc[16] = {};
  for (int i = 0; i < 16; ++i) {
    const int kc = i * 32;
#if USE_TDM
    if (wave == 0 && i < 15)
      tdm_load_wr_panel(WrBf + (size_t)(kc + 32), lds_off((i & 1) ? buf0 : buf1));
    if (i < 15) __builtin_amdgcn_s_wait_tensorcnt(1);
    else        __builtin_amdgcn_s_wait_tensorcnt(0);
    __syncthreads();  // panel i visible to all waves
#endif

    // A fragment: fp32 relation row -> bf16 (one contiguous 16-element k-run per lane)
    v16bf a;
    const float* ap = Abase + (size_t)mRow * E_ + kc + hl * 16;
    v4f x0 = *(const v4f*)(ap);
    v4f x1 = *(const v4f*)(ap + 4);
    v4f x2 = *(const v4f*)(ap + 8);
    v4f x3 = *(const v4f*)(ap + 12);
    if (kc + 32 < E_) __builtin_prefetch(ap + 32, 0, 3);  // near-cache prefetch of next A step
#pragma unroll
    for (int e = 0; e < 4; ++e) {
      a[e] = (__bf16)x0[e]; a[4 + e] = (__bf16)x1[e];
      a[8 + e] = (__bf16)x2[e]; a[12 + e] = (__bf16)x3[e];
    }

#if USE_TDM
    const char* ldsB = (i & 1) ? buf1 : buf0;
    const char* bcol = ldsB + (wx * 256 + ln) * BT_PITCH + hl * 32;
    // software-pipelined B fragments: load block nb+1 while WMMA consumes block nb
    v8bf blo = *(const v8bf*)(bcol);
    v8bf bhi = *(const v8bf*)(bcol + 16);
#pragma unroll
    for (int nb = 0; nb < 16; ++nb) {
      v8bf nlo = blo, nhi = bhi;
      if (nb < 15) {
        const char* bp = bcol + (nb + 1) * (16 * BT_PITCH);
        nlo = *(const v8bf*)(bp);
        nhi = *(const v8bf*)(bp + 16);
      }
      v16bf bf;
#pragma unroll
      for (int e = 0; e < 8; ++e) { bf[e] = blo[e]; bf[8 + e] = bhi[e]; }
      acc[nb] = __builtin_amdgcn_wmma_f32_16x16x32_bf16(false, a, false, bf,
                                                        (short)0, acc[nb], false, false);
      blo = nlo; bhi = nhi;
    }
    __syncthreads();  // read-fence: panel (i&1) may be DMA-retargeted at step i+1
#else
#pragma unroll
    for (int nb = 0; nb < 16; ++nb) {
      const int n = wx * 256 + nb * 16 + ln;
      v16bf bf = *(const v16bf*)(WrBf + (size_t)n * E_ + kc + hl * 16);
      acc[nb] = __builtin_amdgcn_wmma_f32_16x16x32_bf16(false, a, false, bf,
                                                        (short)0, acc[nb], false, false);
    }
#endif
  }

  // stage rel tile to LDS (bf16), aliasing panel 0 (all DMA complete here)
  __bf16* relLds = (__bf16*)smem;
#pragma unroll
  for (int nb = 0; nb < 16; ++nb) {
    const int n = wx * 256 + nb * 16 + ln;
#pragma unroll
    for (int r = 0; r < 8; ++r) {
      const int m = wy * 16 + r + hl * 8;
      relLds[m * RELPITCH + n] = (__bf16)acc[nb][r];
    }
  }
  __syncthreads();

  // combine: one thread per (s,h)
  const int s = tid >> 3;   // 0..31
  const int h = tid & 7;
  const float* qrow = qf + ((size_t)b * L_ + sBase + s) * E_ + h * D_;
  const float* krow = kf + ((size_t)b * L_ + t) * E_ + h * D_;
  const __bf16* rqp = relLds + s * RELPITCH + h * D_;
  const __bf16* rkp = rqp + E_;
  float dot = 0.0f;
#pragma unroll 8
  for (int n = 0; n < D_; ++n)
    dot += (qrow[n] + (float)rqp[n]) * (krow[n] + (float)rkp[n]);
  attn[(((size_t)b * L_ + sBase + s) * L_ + t) * H_ + h] = dot;
}

// ---------------------------------------------------------------- masked softmax over key axis
__global__ __launch_bounds__(256) void masked_softmax(const int* __restrict__ adj,
                                                      float* __restrict__ attn) {
  const int wave = threadIdx.x >> 5;
  const int lane = threadIdx.x & 31;
  int row = blockIdx.x * 8 + wave;   // over B*L*H
  const int h = row & 7;   row >>= 3;
  const int s = row & 255; row >>= 8;
  const int b = row;
  float* base = attn + (((size_t)b * L_ + s) * L_) * H_ + h;
  const int* mrow = adj + ((size_t)b * L_ + s) * L_;

  float vals[8];
  float mx = -__builtin_inff();
#pragma unroll
  for (int i = 0; i < 8; ++i) {
    const int t = lane + 32 * i;
    float x = base[(size_t)t * H_];
    if (mrow[t] == 1) x = -__builtin_inff();
    vals[i] = x;
    mx = fmaxf(mx, x);
  }
#pragma unroll
  for (int o = 16; o > 0; o >>= 1) mx = fmaxf(mx, __shfl_xor(mx, o, 32));
  float sum = 0.0f;
#pragma unroll
  for (int i = 0; i < 8; ++i) { vals[i] = __expf(vals[i] - mx); sum += vals[i]; }
#pragma unroll
  for (int o = 16; o > 0; o >>= 1) sum += __shfl_xor(sum, o, 32);
  const float inv = 1.0f / sum;
#pragma unroll
  for (int i = 0; i < 8; ++i) base[(size_t)(lane + 32 * i) * H_] = vals[i] * inv;
}

// ---------------------------------------------------------------- AV: out[b,s,h,d] = sum_t w*v
__global__ __launch_bounds__(256) void attn_v(const float* __restrict__ attnW,
                                              const float* __restrict__ vf,
                                              float* __restrict__ av) {
  __shared__ float wrow[L_ * H_];  // 8 KB
  const int b = blockIdx.x >> 8;
  const int s = blockIdx.x & 255;
  const float* wsrc = attnW + (((size_t)b * L_ + s) * L_) * H_;
  for (int i = threadIdx.x; i < L_ * H_; i += 256) wrow[i] = wsrc[i];
  __syncthreads();
#pragma unroll
  for (int rep = 0; rep < 2; ++rep) {
    const int idx = threadIdx.x + rep * 256;  // 0..511 = h*64+d
    const int h = idx >> 6;
    const float* vcol = vf + (size_t)b * L_ * E_ + idx;
    float acc = 0.0f;
    for (int t = 0; t < L_; ++t) acc += wrow[t * H_ + h] * vcol[(size_t)t * E_];
    av[((size_t)b * L_ + s) * E_ + idx] = acc;
  }
}

// ---------------------------------------------------------------- host orchestration
extern "C" void kernel_launch(void* const* d_in, const int* in_sizes, int n_in,
                              void* d_out, int out_size, void* d_ws, size_t ws_size,
                              hipStream_t stream) {
  (void)in_sizes; (void)n_in; (void)out_size; (void)ws_size;
  const float* queries  = (const float*)d_in[0];
  const float* keys     = (const float*)d_in[1];
  const float* values   = (const float*)d_in[2];
  const float* relation = (const float*)d_in[3];
  const int*   adj      = (const int*)d_in[4];
  const float* Wq = (const float*)d_in[5];
  const float* bq = (const float*)d_in[6];
  const float* Wk = (const float*)d_in[7];
  const float* bk = (const float*)d_in[8];
  const float* Wv = (const float*)d_in[9];
  const float* bv = (const float*)d_in[10];
  const float* Wr = (const float*)d_in[11];
  const float* Wo = (const float*)d_in[12];
  const float* bo = (const float*)d_in[13];

  const int NE  = E_ * E_;        // 262144
  const int NX  = B_ * L_ * E_;   // 262144
  const int NWr = TWOE_ * E_;     // 524288

  char* p = (char*)d_ws;
  auto carve = [&](size_t bytes) { char* r = p; p += (bytes + 255) & ~(size_t)255; return r; };
  __bf16* WqBf = (__bf16*)carve((size_t)NE * 2);
  __bf16* WkBf = (__bf16*)carve((size_t)NE * 2);
  __bf16* WvBf = (__bf16*)carve((size_t)NE * 2);
  __bf16* WoBf = (__bf16*)carve((size_t)NE * 2);
  __bf16* WrBf = (__bf16*)carve((size_t)NWr * 2);
  __bf16* XqBf = (__bf16*)carve((size_t)NX * 2);
  __bf16* XkBf = (__bf16*)carve((size_t)NX * 2);
  __bf16* XvBf = (__bf16*)carve((size_t)NX * 2);
  float*  qF   = (float*)carve((size_t)NX * 4);
  float*  kF   = (float*)carve((size_t)NX * 4);
  float*  vF   = (float*)carve((size_t)NX * 4);
  float*  avF  = (float*)carve((size_t)NX * 4);

  float* outp  = (float*)d_out;                // [B,L,E]
  float* attnp = outp + (size_t)B_ * L_ * E_;  // [B,L,L,H]

  cvt_f32_bf16<<<(NE  + 255) / 256, 256, 0, stream>>>(Wq, WqBf, NE);
  cvt_f32_bf16<<<(NE  + 255) / 256, 256, 0, stream>>>(Wk, WkBf, NE);
  cvt_f32_bf16<<<(NE  + 255) / 256, 256, 0, stream>>>(Wv, WvBf, NE);
  cvt_f32_bf16<<<(NE  + 255) / 256, 256, 0, stream>>>(Wo, WoBf, NE);
  cvt_f32_bf16<<<(NWr + 255) / 256, 256, 0, stream>>>(Wr, WrBf, NWr);
  cvt_f32_bf16<<<(NX  + 255) / 256, 256, 0, stream>>>(queries, XqBf, NX);
  cvt_f32_bf16<<<(NX  + 255) / 256, 256, 0, stream>>>(keys,    XkBf, NX);
  cvt_f32_bf16<<<(NX  + 255) / 256, 256, 0, stream>>>(values,  XvBf, NX);

  gemm_bias<false><<<(B_ * L_) / 16, 256, 0, stream>>>(XqBf, WqBf, bq, qF,
                                                       B_ * L_, E_, E_, SCALE_);
  gemm_bias<false><<<(B_ * L_) / 16, 256, 0, stream>>>(XkBf, WkBf, bk, kF,
                                                       B_ * L_, E_, E_, 1.0f);
  gemm_bias<false><<<(B_ * L_) / 16, 256, 0, stream>>>(XvBf, WvBf, bv, vF,
                                                       B_ * L_, E_, E_, 1.0f);

  rel_attn_logits<<<B_ * L_ * (L_ / 32), 256, 0, stream>>>(relation, WrBf, qF, kF, attnp);

  masked_softmax<<<(B_ * L_ * H_) / 8, 256, 0, stream>>>(adj, attnp);

  attn_v<<<B_ * L_, 256, 0, stream>>>(attnp, vF, avF);

  gemm_bias<true><<<(B_ * L_) / 16, 256, 0, stream>>>(avF, WoBf, bo, outp,
                                                      B_ * L_, E_, E_, 1.0f);
}
